// tinystoriesgru_23115513987273
// MI455X (gfx1250) — matrix operand order repaired and checked by reading, not compiled, not run on hardware
//
#include <hip/hip_runtime.h>

// ---------------- types ----------------
typedef __bf16 bf16_t;
typedef __attribute__((ext_vector_type(8)))  __bf16 v8bf;
typedef __attribute__((ext_vector_type(16))) __bf16 v16bf;
typedef __attribute__((ext_vector_type(8)))  float  v8f;

// ---------------- model dims ----------------
constexpr int kV = 128, kE = 384, kH = 6, kD = 64, kT = 256, kB = 64;
constexpr int kG = 3 * kE;            // 1152
constexpr int BT = kB * kT;           // 16384

// ---------------- workspace layout (bytes, all 256-aligned) ----------------
constexpr size_t SZ_BF   = (size_t)BT * kE * 2;              // 12,582,912
constexpr size_t OFF_TOK = 0;
constexpr size_t OFF_CTX2= OFF_TOK + SZ_BF;
constexpr size_t OFF_RNN = OFF_CTX2 + SZ_BF;                 // f32 BT*E
constexpr size_t OFF_W   = OFF_RNN + (size_t)BT * kE * 4;
constexpr size_t W_INPROJ= OFF_W;
constexpr size_t W_OUTPROJ = W_INPROJ + (size_t)kG*kE*2;
constexpr size_t W_TOK   = W_OUTPROJ + (size_t)kE*kE*2;
constexpr size_t W_CTX   = W_TOK + (size_t)kG*kE*2;
constexpr size_t W_MEM   = W_CTX + (size_t)kG*kE*2;
constexpr size_t W_HH    = W_MEM + (size_t)kG*kE*2;
constexpr size_t W_MW    = W_HH  + (size_t)kG*kE*2;
constexpr size_t W_HEAD  = W_MW  + (size_t)kE*kE*2;
constexpr size_t OFF_U   = W_HEAD + (size_t)kV*kE*2;
// union region U (75.5 MB): lifetimes do not overlap
constexpr size_t U_X   = OFF_U;                // bf16 x, dead after qkv gemm
constexpr size_t U_Q   = OFF_U + 1*SZ_BF;      // dead after attention
constexpr size_t U_K   = OFF_U + 2*SZ_BF;
constexpr size_t U_VT  = OFF_U + 3*SZ_BF;
constexpr size_t U_CTX = OFF_U + 4*SZ_BF;      // dead after out_proj
constexpr size_t U_PRE = OFF_U;                // f32 (T,B,G), dead after scan
constexpr size_t U_NORM= OFF_U;                // bf16, after scan

// ---------------- WMMA helpers ----------------
__device__ __forceinline__ v8f wmma_bf16(v16bf a, v16bf b, v8f c) {
  return __builtin_amdgcn_wmma_f32_16x16x32_bf16(false, a, false, b, (short)0, c,
                                                 false, false);
}

// A-operand (16xK, MxK): ISA 7.12.2 layout. lane<16 -> K{0..7,16..23}, lane>=16 -> K{8..15,24..31}
__device__ __forceinline__ v16bf load_a(const bf16_t* base, int ld, int lane) {
  int row = lane & 15;
  const bf16_t* p = base + (size_t)row * ld + ((lane < 16) ? 0 : 8);
  v8bf lo = *(const v8bf*)p;
  v8bf hi = *(const v8bf*)(p + 16);
  v16bf r;
#pragma unroll
  for (int i = 0; i < 8; i++) { r[i] = lo[i]; r[8 + i] = hi[i]; }
  return r;
}

// B-operand (KxN): lane n holds column n = row n of the NT weight matrix;
// half-wave split along K: lane<16 -> K{0..7,16..23}, lane>=16 -> K{8..15,24..31} (matches A).
__device__ __forceinline__ v16bf load_b(const bf16_t* base, int ld, int lane) {
  int row = lane & 15;
  const bf16_t* p = base + (size_t)row * ld + ((lane < 16) ? 0 : 8);
  v8bf lo = *(const v8bf*)p;
  v8bf hi = *(const v8bf*)(p + 16);
  v16bf r;
#pragma unroll
  for (int i = 0; i < 8; i++) { r[i] = lo[i]; r[8 + i] = hi[i]; }
  return r;
}

__device__ __forceinline__ float sigmoidf(float x) {
  return 1.0f / (1.0f + __expf(-x));
}

// Async global->LDS copy of one 16-byte chunk (GLOBAL_LOAD_ASYNC_TO_LDS_B128,
// ASYNCcnt-tracked). ldsaddr = LDS byte address (generic addr truncated to 32b
// per the aperture mapping), gaddr = 64-bit global address.
__device__ __forceinline__ void async_copy_b128(unsigned int ldsaddr,
                                                unsigned long long gaddr) {
  asm volatile("global_load_async_to_lds_b128 %0, %1, off"
               :
               : "v"(ldsaddr), "v"(gaddr)
               : "memory");
}
__device__ __forceinline__ void async_wait0() {
  asm volatile("s_wait_asynccnt 0" ::: "memory");
}

// ---------------- weight conversion ----------------
__global__ void cvt_f32_bf16(const float* __restrict__ s, bf16_t* __restrict__ d, int n) {
  int i = blockIdx.x * blockDim.x + threadIdx.x;
  if (i < n) d[i] = (bf16_t)s[i];
}
__global__ void cvt_slice_bf16(const float* __restrict__ s, bf16_t* __restrict__ d,
                               int rows, int cols, int ld, int off) {
  int i = blockIdx.x * blockDim.x + threadIdx.x;
  if (i < rows * cols) {
    int r = i / cols, c = i - r * cols;
    d[i] = (bf16_t)s[(size_t)r * ld + off + c];
  }
}

// ---------------- embedding ----------------
__global__ void embed_kernel(const int* __restrict__ idx, const float* __restrict__ te,
                             const float* __restrict__ pe, bf16_t* __restrict__ xbf,
                             bf16_t* __restrict__ tokbf) {
  int r = blockIdx.x;          // 0..BT-1
  int e = threadIdx.x;         // 0..383
  int t = r & (kT - 1);
  int tokid = idx[r];
  float tv = te[(size_t)tokid * kE + e];
  tokbf[(size_t)r * kE + e] = (bf16_t)tv;
  xbf[(size_t)r * kE + e] = (bf16_t)(tv + pe[(size_t)t * kE + e]);
}

// ---------------- generic NT GEMM (bf16 x bf16 -> f32 acc) ----------------
// C[M,N] = A[M,K] * W[N,K]^T + bias ; MODE: 0 row-major f32, 1 row-major bf16,
// 2 f32 at (t*B+b, g) ("pre" layout), 3 scatter into q/k/vt head layouts.
// The 16-row A strip (contiguous 16*K bf16, K<=384) is staged into LDS with
// async loads and shared by all 8 waves of the block.
template <int MODE, bool ACCUM>
__global__ __launch_bounds__(256) void gemm_nt(
    const bf16_t* __restrict__ A, const bf16_t* __restrict__ W,
    const float* __restrict__ bias, float* __restrict__ Cf, bf16_t* __restrict__ Cb,
    bf16_t* __restrict__ Qp, bf16_t* __restrict__ Kp, bf16_t* __restrict__ Vtp,
    int K, int N) {
  __shared__ __align__(16) bf16_t As[16 * 384];
  const int wave = threadIdx.x >> 5;
  const int lane = threadIdx.x & 31;
  const int r0 = blockIdx.x * 16;
  const int n0 = (blockIdx.y * 8 + wave) * 16;
  const bf16_t* arow = A + (size_t)r0 * K;
  const bf16_t* wrow = W + (size_t)n0 * K;

  // async-stage A strip: 16*K*2 bytes in 16B chunks
  {
    unsigned int lbase = (unsigned int)(size_t)(void*)As;
    int nchunks = (16 * K * 2) >> 4;
    for (int c = threadIdx.x; c < nchunks; c += 256) {
      unsigned int off = (unsigned int)c << 4;
      async_copy_b128(lbase + off, (unsigned long long)((const char*)arow + off));
    }
    async_wait0();
  }
  __syncthreads();

  v8f acc = {};
  for (int k0 = 0; k0 < K; k0 += 32) {
    v16bf a = load_a(As + k0, K, lane);
    v16bf b = load_b(wrow + k0, K, lane);
    if (k0 + 32 < K)
      __builtin_prefetch(wrow + (size_t)(lane & 15) * K + k0 + 32, 0, 1);
    acc = wmma_bf16(a, b, acc);
  }
  const int col = lane & 15;
  const int rb = (lane < 16) ? 0 : 8;
  const float bv = bias ? bias[n0 + col] : 0.0f;
#pragma unroll
  for (int i = 0; i < 8; i++) {
    int r = r0 + rb + i;
    float v = acc[i] + bv;
    if (MODE == 0) {
      size_t o = (size_t)r * N + n0 + col;
      if (ACCUM) v += Cf[o];
      Cf[o] = v;
    } else if (MODE == 1) {
      Cb[(size_t)r * N + n0 + col] = (bf16_t)v;
    } else if (MODE == 2) {
      int b = r >> 8, t = r & (kT - 1);                 // r = b*T + t
      size_t o = ((size_t)t * kB + b) * kG + n0 + col;  // (T,B,G) layout
      if (ACCUM) v += Cf[o];
      Cf[o] = v;
    } else {  // QKV scatter
      int b = r >> 8, t = r & (kT - 1);
      int g = n0 + col;
      int sec = g / kE, hd = g % kE;
      int h = hd >> 6, d = hd & 63;
      size_t bh = (size_t)b * kH + h;
      if (sec == 0)      Qp[(bh * kT + t) * kD + d] = (bf16_t)v;
      else if (sec == 1) Kp[(bh * kT + t) * kD + d] = (bf16_t)v;
      else               Vtp[(bh * kD + d) * kT + t] = (bf16_t)v;  // transposed V
    }
  }
}

// ---------------- causal attention (one (b,h) per block, 4 waves) ----------------
// K panel (256x64 bf16 = 32 KB) is async-staged to LDS and shared by the waves.
__global__ __launch_bounds__(128) void attn_kernel(const bf16_t* __restrict__ q,
                                                   const bf16_t* __restrict__ k,
                                                   const bf16_t* __restrict__ vt,
                                                   bf16_t* __restrict__ ctx) {
  __shared__ __align__(16) float  Ssc[4][16 * 256];
  __shared__ __align__(16) bf16_t Pm[4][16 * 256];
  __shared__ __align__(16) bf16_t Ks[kT * kD];   // 32 KB
  const int bh = blockIdx.x;
  const int b = bh / kH, h = bh % kH;
  const int wave = threadIdx.x >> 5, lane = threadIdx.x & 31;
  const bf16_t* qp = q + (size_t)bh * kT * kD;
  const bf16_t* kp = k + (size_t)bh * kT * kD;
  const bf16_t* vtp = vt + (size_t)bh * kD * kT;
  float* S = Ssc[wave];
  bf16_t* P = Pm[wave];
  const int col = lane & 15, rb = (lane < 16) ? 0 : 8;

  // async-stage K panel: 32768 bytes in 16B chunks (2048 chunks / 128 threads)
  {
    unsigned int lbase = (unsigned int)(size_t)(void*)Ks;
    for (int c = threadIdx.x; c < (kT * kD * 2) >> 4; c += 128) {
      unsigned int off = (unsigned int)c << 4;
      async_copy_b128(lbase + off, (unsigned long long)((const char*)kp + off));
    }
    async_wait0();
  }
  __syncthreads();

  for (int qi = wave; qi < 16; qi += 4) {
    const int r0 = qi * 16;
    v16bf a0 = load_a(qp + (size_t)r0 * kD, kD, lane);
    v16bf a1 = load_a(qp + (size_t)r0 * kD + 32, kD, lane);
    // scores S = (Q Kt) * 1/8 with causal mask, staged to LDS
    for (int kj = 0; kj <= qi; kj++) {
      v8f acc = {};
      v16bf b0 = load_b(Ks + (size_t)kj * 16 * kD, kD, lane);
      acc = wmma_bf16(a0, b0, acc);
      v16bf b1 = load_b(Ks + (size_t)kj * 16 * kD + 32, kD, lane);
      acc = wmma_bf16(a1, b1, acc);
#pragma unroll
      for (int i = 0; i < 8; i++) {
        int rr = r0 + rb + i, cc = kj * 16 + col;
        float s = acc[i] * 0.125f;
        if (cc > rr) s = -1e30f;
        S[(rb + i) * 256 + cc] = s;
      }
    }
    const int nks = (qi + 2) >> 1;  // #32-wide k-steps covering valid keys
    // row softmax (lanes 0..15, one row each)
    if (lane < 16) {
      int rr = lane, len = r0 + rr + 1;
      float mx = -1e30f;
      for (int j = 0; j < len; j++) mx = fmaxf(mx, S[rr * 256 + j]);
      float sum = 0.0f;
      for (int j = 0; j < len; j++) sum += __expf(S[rr * 256 + j] - mx);
      float inv = 1.0f / sum;
      for (int j = 0; j < len; j++)
        P[rr * 256 + j] = (bf16_t)(__expf(S[rr * 256 + j] - mx) * inv);
      for (int j = len; j < nks * 32; j++) P[rr * 256 + j] = (bf16_t)0.0f;
    }
    // ctx = P @ V  (B-operand = rows of Vt, K-contiguous)
    for (int n = 0; n < 4; n++) {
      v8f acc = {};
      for (int ks = 0; ks < nks; ks++) {
        v16bf a = load_a(P + ks * 32, 256, lane);
        v16bf bb = load_b(vtp + (size_t)n * 16 * kT + ks * 32, kT, lane);
        acc = wmma_bf16(a, bb, acc);
      }
#pragma unroll
      for (int i = 0; i < 8; i++) {
        int rr = r0 + rb + i;
        ctx[((size_t)b * kT + rr) * kE + h * 64 + n * 16 + col] = (bf16_t)acc[i];
      }
    }
  }
}

// ---------------- persistent GRU-memory scan: 4 blocks x 16 batch rows ----------------
__global__ __launch_bounds__(256) void scan_kernel(
    const float* __restrict__ pre, const bf16_t* __restrict__ w_mem,
    const bf16_t* __restrict__ w_hh, const float* __restrict__ b_hh,
    const float* __restrict__ pg_w, const float* __restrict__ pg_b,
    const bf16_t* __restrict__ mw, const float* __restrict__ mw_b,
    float* __restrict__ rnn) {
  __shared__ __align__(16) float  hs[16 * kE];
  __shared__ __align__(16) float  ms[16 * kE];
  __shared__ __align__(16) bf16_t hb[16 * kE];
  __shared__ __align__(16) bf16_t mb[16 * kE];
  __shared__ __align__(16) bf16_t gi[16 * kG];
  __shared__ __align__(16) bf16_t gh[16 * kG];
  __shared__ float pvec[16];
  const int b0 = blockIdx.x * 16;
  const int wave = threadIdx.x >> 5, lane = threadIdx.x & 31;
  const int col = lane & 15, rb = (lane < 16) ? 0 : 8;

  for (int i = threadIdx.x; i < 16 * kE; i += 256) {
    hs[i] = 0.0f; ms[i] = 0.0f;
    hb[i] = (bf16_t)0.0f; mb[i] = (bf16_t)0.0f;
  }
  __syncthreads();

  for (int t = 0; t < kT; t++) {
    // phase A: gi = pre_t + m @ Wmem^T ; gh = h @ Whh^T + b_hh   (144 tiles / 8 waves)
    for (int w = 0; w < 18; w++) {
      int idx = wave * 18 + w;
      bool isGi = idx < 72;
      int g0 = (isGi ? idx : idx - 72) * 16;
      const bf16_t* Wm = isGi ? w_mem : w_hh;
      const bf16_t* Ab = isGi ? mb : hb;
      v8f acc = {};
#pragma unroll
      for (int k0 = 0; k0 < kE; k0 += 32) {
        v16bf a = load_a(Ab + k0, kE, lane);
        v16bf bb = load_b(Wm + (size_t)g0 * kE + k0, kE, lane);
        if (k0 + 32 < kE)
          __builtin_prefetch(Wm + (size_t)(g0 + (lane & 15)) * kE + k0 + 32, 0, 1);
        acc = wmma_bf16(a, bb, acc);
      }
      if (isGi) {
#pragma unroll
        for (int i = 0; i < 8; i++) {
          size_t o = ((size_t)t * kB + (b0 + rb + i)) * kG + g0 + col;
          gi[(rb + i) * kG + g0 + col] = (bf16_t)(acc[i] + pre[o]);
        }
      } else {
        float bv = b_hh[g0 + col];
#pragma unroll
        for (int i = 0; i < 8; i++)
          gh[(rb + i) * kG + g0 + col] = (bf16_t)(acc[i] + bv);
      }
    }
    __syncthreads();
    // phase B: gates + h update, emit rnn_out
    for (int i = threadIdx.x; i < 16 * kE; i += 256) {
      int r = i / kE, e = i - r * kE;
      float ir = (float)gi[r * kG + e];
      float iz = (float)gi[r * kG + kE + e];
      float in_ = (float)gi[r * kG + 2 * kE + e];
      float hr = (float)gh[r * kG + e];
      float hz = (float)gh[r * kG + kE + e];
      float hn = (float)gh[r * kG + 2 * kE + e];
      float rg = sigmoidf(ir + hr);
      float z = sigmoidf(iz + hz);
      float n = tanhf(in_ + rg * hn);
      float hnew = (1.0f - z) * n + z * hs[i];
      hs[i] = hnew;
      hb[i] = (bf16_t)hnew;
      rnn[((size_t)(b0 + r) * kT + t) * kE + e] = hnew;
    }
    __syncthreads();
    // phase C: write-gate scalar per row
    if (threadIdx.x < 16) {
      float d = 0.0f;
      for (int e = 0; e < kE; e++) d += hs[threadIdx.x * kE + e] * pg_w[e];
      pvec[threadIdx.x] = sigmoidf(d + pg_b[0]);
    }
    __syncthreads();
    // phase D: cand = tanh(h_new @ mw^T + b); m update   (24 tiles / 8 waves)
    for (int w = 0; w < 3; w++) {
      int e0 = (wave * 3 + w) * 16;
      v8f acc = {};
#pragma unroll
      for (int k0 = 0; k0 < kE; k0 += 32) {
        v16bf a = load_a(hb + k0, kE, lane);
        v16bf bb = load_b(mw + (size_t)e0 * kE + k0, kE, lane);
        acc = wmma_bf16(a, bb, acc);
      }
      float bv = mw_b[e0 + col];
#pragma unroll
      for (int i = 0; i < 8; i++) {
        int r = rb + i;
        float cand = tanhf(acc[i] + bv);
        float p = pvec[r];
        int o = r * kE + e0 + col;
        float mn = (1.0f - p) * ms[o] + p * cand;
        ms[o] = mn;
        mb[o] = (bf16_t)mn;
      }
    }
    __syncthreads();
  }
}

// ---------------- LayerNorm -> bf16 ----------------
__global__ __launch_bounds__(128) void ln_kernel(const float* __restrict__ rnn,
                                                 const float* __restrict__ g,
                                                 const float* __restrict__ bb,
                                                 bf16_t* __restrict__ out) {
  __shared__ float red[128];
  int r = blockIdx.x;
  const float* x = rnn + (size_t)r * kE;
  float s = 0.0f;
  for (int e = threadIdx.x; e < kE; e += 128) s += x[e];
  red[threadIdx.x] = s;
  __syncthreads();
  for (int st = 64; st > 0; st >>= 1) {
    if (threadIdx.x < st) red[threadIdx.x] += red[threadIdx.x + st];
    __syncthreads();
  }
  float mu = red[0] / kE;
  __syncthreads();
  float v = 0.0f;
  for (int e = threadIdx.x; e < kE; e += 128) { float d = x[e] - mu; v += d * d; }
  red[threadIdx.x] = v;
  __syncthreads();
  for (int st = 64; st > 0; st >>= 1) {
    if (threadIdx.x < st) red[threadIdx.x] += red[threadIdx.x + st];
    __syncthreads();
  }
  float inv = rsqrtf(red[0] / kE + 1e-5f);
  for (int e = threadIdx.x; e < kE; e += 128)
    out[(size_t)r * kE + e] = (bf16_t)((x[e] - mu) * inv * g[e] + bb[e]);
}

// ---------------- host launch ----------------
extern "C" void kernel_launch(void* const* d_in, const int* in_sizes, int n_in,
                              void* d_out, int out_size, void* d_ws, size_t ws_size,
                              hipStream_t stream) {
  const int*   idx       = (const int*)d_in[0];
  const float* token_emb = (const float*)d_in[1];
  const float* pos_emb   = (const float*)d_in[2];
  const float* in_proj_w = (const float*)d_in[3];
  const float* in_proj_b = (const float*)d_in[4];
  const float* out_proj_w= (const float*)d_in[5];
  const float* out_proj_b= (const float*)d_in[6];
  const float* w_ih      = (const float*)d_in[7];
  const float* b_ih      = (const float*)d_in[8];
  const float* w_hh      = (const float*)d_in[9];
  const float* b_hh      = (const float*)d_in[10];
  const float* pg_w      = (const float*)d_in[11];
  const float* pg_b      = (const float*)d_in[12];
  const float* mw_w      = (const float*)d_in[13];
  const float* mw_b      = (const float*)d_in[14];
  const float* ln_g      = (const float*)d_in[15];
  const float* ln_b      = (const float*)d_in[16];
  const float* head_w    = (const float*)d_in[17];
  const float* head_b    = (const float*)d_in[18];

  char* ws = (char*)d_ws;
  bf16_t* tokbf   = (bf16_t*)(ws + OFF_TOK);
  bf16_t* ctx2bf  = (bf16_t*)(ws + OFF_CTX2);
  float*  rnn     = (float*)(ws + OFF_RNN);
  bf16_t* inprojB = (bf16_t*)(ws + W_INPROJ);
  bf16_t* outprojB= (bf16_t*)(ws + W_OUTPROJ);
  bf16_t* wtokB   = (bf16_t*)(ws + W_TOK);
  bf16_t* wctxB   = (bf16_t*)(ws + W_CTX);
  bf16_t* wmemB   = (bf16_t*)(ws + W_MEM);
  bf16_t* whhB    = (bf16_t*)(ws + W_HH);
  bf16_t* mwB     = (bf16_t*)(ws + W_MW);
  bf16_t* headB   = (bf16_t*)(ws + W_HEAD);
  bf16_t* xbf     = (bf16_t*)(ws + U_X);
  bf16_t* qbf     = (bf16_t*)(ws + U_Q);
  bf16_t* kbf     = (bf16_t*)(ws + U_K);
  bf16_t* vtbf    = (bf16_t*)(ws + U_VT);
  bf16_t* ctxbf   = (bf16_t*)(ws + U_CTX);
  float*  pre     = (float*)(ws + U_PRE);
  bf16_t* normbf  = (bf16_t*)(ws + U_NORM);
  float*  logits  = (float*)d_out;

  // weight conversions
  cvt_f32_bf16<<<(kG * kE + 255) / 256, 256, 0, stream>>>(in_proj_w, inprojB, kG * kE);
  cvt_f32_bf16<<<(kE * kE + 255) / 256, 256, 0, stream>>>(out_proj_w, outprojB, kE * kE);
  cvt_slice_bf16<<<(kG * kE + 255) / 256, 256, 0, stream>>>(w_ih, wtokB, kG, kE, kG, 0);
  cvt_slice_bf16<<<(kG * kE + 255) / 256, 256, 0, stream>>>(w_ih, wctxB, kG, kE, kG, kE);
  cvt_slice_bf16<<<(kG * kE + 255) / 256, 256, 0, stream>>>(w_ih, wmemB, kG, kE, kG, 2 * kE);
  cvt_f32_bf16<<<(kG * kE + 255) / 256, 256, 0, stream>>>(w_hh, whhB, kG * kE);
  cvt_f32_bf16<<<(kE * kE + 255) / 256, 256, 0, stream>>>(mw_w, mwB, kE * kE);
  cvt_f32_bf16<<<(kV * kE + 255) / 256, 256, 0, stream>>>(head_w, headB, kV * kE);

  // embed
  embed_kernel<<<BT, kE, 0, stream>>>(idx, token_emb, pos_emb, xbf, tokbf);

  // qkv projection, scatter to head layouts (V transposed)
  gemm_nt<3, false><<<dim3(BT / 16, kG / 128), 256, 0, stream>>>(
      xbf, inprojB, in_proj_b, nullptr, nullptr, qbf, kbf, vtbf, kE, kG);

  // causal attention
  attn_kernel<<<kB * kH, 128, 0, stream>>>(qbf, kbf, vtbf, ctxbf);

  // out projection
  gemm_nt<1, false><<<dim3(BT / 16, kE / 128), 256, 0, stream>>>(
      ctxbf, outprojB, out_proj_b, nullptr, ctx2bf, nullptr, nullptr, nullptr, kE, kE);

  // pre = tok @ w_tok^T + ctx2 @ w_ctx^T + b_ih   (stored (T,B,G)-major)
  gemm_nt<2, false><<<dim3(BT / 16, kG / 128), 256, 0, stream>>>(
      tokbf, wtokB, b_ih, pre, nullptr, nullptr, nullptr, nullptr, kE, kG);
  gemm_nt<2, true><<<dim3(BT / 16, kG / 128), 256, 0, stream>>>(
      ctx2bf, wctxB, nullptr, pre, nullptr, nullptr, nullptr, nullptr, kE, kG);

  // recurrent scan
  scan_kernel<<<4, 256, 0, stream>>>(pre, wmemB, whhB, b_hh, pg_w, pg_b, mwB, mw_b, rnn);

  // layernorm + head
  ln_kernel<<<BT, 128, 0, stream>>>(rnn, ln_g, ln_b, normbf);
  gemm_nt<0, false><<<dim3(BT / 16, kV / 128), 256, 0, stream>>>(
      normbf, headB, head_b, logits, nullptr, nullptr, nullptr, nullptr, kE, kV);

  (void)in_sizes; (void)n_in; (void)out_size; (void)ws_size;
}
